// KnowledgeEncoder_18940805775696
// MI455X (gfx1250) — compile-verified, matrix operand order checked by reading, not run on hardware
//
#include <hip/hip_runtime.h>
#include <hip/hip_bf16.h>

// ---------------------------------------------------------------------------
// Problem constants (from reference): E=HID=128, B=8, N=8, L=32, K=256, R=8, NB=4
// ---------------------------------------------------------------------------
#define EB   128
#define BB   8
#define NN_  8
#define LL   32
#define KK   256
#define RR   8
#define NBB  4
#define BN   (BB*NN_)      // 64
#define BK   (BB*KK)       // 2048
#define FIN1_LD 288        // 257 padded to mult of 32
#define FIN2_LD 544        // 513 padded to mult of 32

// RGCN dynamic LDS layout (bytes): sX 64KB | sXr 64KB | sG 128KB  = 256KB
#define RGCN_SX_OFF   0
#define RGCN_SXR_OFF  65536
#define RGCN_SG_OFF   131072
#define RGCN_LDS_SIZE 262144

typedef __attribute__((ext_vector_type(16))) __bf16 v16bf;
typedef __attribute__((ext_vector_type(8)))  __bf16 v8bf;
typedef __attribute__((ext_vector_type(8)))  float  v8f;

#define DEV __device__ __forceinline__

// Load a 16x32 bf16 fragment (A layout; also B when source is stored N-major,
// i.e. B^T row-major).  Per-lane: outer index = lane&15, K mapping
// K = (e&7) + 8*(lane>>4) + 16*(e>>3)  -> two contiguous 16B loads per lane.
DEV v16bf load_frag(const __bf16* p, int ld, int lane) {
  const int o = lane & 15, g = lane >> 4;
  const __bf16* r = p + (size_t)o * ld + 8 * g;
  v8bf lo = *(const v8bf*)(r);
  v8bf hi = *(const v8bf*)(r + 16);
  v16bf f;
#pragma unroll
  for (int i = 0; i < 8; ++i) { f[i] = lo[i]; f[i + 8] = hi[i]; }
  return f;
}

DEV v8f wmma_bf16(v16bf a, v16bf b, v8f c) {
  return __builtin_amdgcn_wmma_f32_16x16x32_bf16(false, a, false, b, (short)0, c,
                                                 false, false);
}

// CDNA5 async copy: 16B global -> LDS, tracked with ASYNCcnt (per-wave,
// in-order retirement).  lds_byte = LDS byte address, g = global address.
DEV void async_copy_b128(unsigned lds_byte, const void* g) {
  asm volatile("global_load_async_to_lds_b128 %0, %1, off"
               :
               : "v"(lds_byte), "v"(g)
               : "memory");
}
#define WAIT_ASYNC_ALL()  asm volatile("s_wait_asynccnt 0x0" ::: "memory")
#define WAIT_ASYNC_16()   asm volatile("s_wait_asynccnt 0x10" ::: "memory")

// ---------------------------------------------------------------------------
// Generic GEMM: C(MxN) f32 = A(MxKd) bf16 row-major  x  B (Bt = B^T, NxKd bf16
// row-major) + bias, one wave per 16x16 tile.  grid = (N/16, M/16), block=32.
// ---------------------------------------------------------------------------
__global__ __launch_bounds__(32)
void gemm_bf16_kernel(const __bf16* __restrict__ A, int lda,
                      const __bf16* __restrict__ Bt, int ldb,
                      const float* __restrict__ bias,
                      float* __restrict__ C, int ldc, int Kd) {
  const int lane = threadIdx.x;
  const int row0 = blockIdx.y * 16, col0 = blockIdx.x * 16;
  v8f acc = {};
  const __bf16* Ab = A + (size_t)row0 * lda;
  const __bf16* Bb = Bt + (size_t)col0 * ldb;
  for (int k0 = 0; k0 < Kd; k0 += 32) {
    v16bf af = load_frag(Ab + k0, lda, lane);
    v16bf bf = load_frag(Bb + k0, ldb, lane);
    acc = wmma_bf16(af, bf, acc);
  }
  const int n_ = lane & 15, g = lane >> 4;
  const float bv = bias ? bias[col0 + n_] : 0.f;
#pragma unroll
  for (int e = 0; e < 8; ++e)
    C[(size_t)(row0 + e + 8 * g) * ldc + col0 + n_] = acc[e] + bv;
}

// ---------------------------------------------------------------------------
// Fused RGCN:  out = relu( sum_r gN[b,r] @ (X @ Wr[r]) + X @ Wself )
// X (256x128 bf16) staged once into LDS via async copy; each gN[b,r] slab
// (256x256 bf16, 128KB) prefetched into LDS asynchronously while stage-1
// WMMAs run.  Xr^T staged in LDS.  block = 512 threads (16 waves), each wave
// owns 8 of the 128 output tiles (wave w -> output row-tile w).
// mode 1: write bf16 into kb_o1 (cols 128:256) and fin2 (cols 128:256)
// mode 2: write f32 into d_out (cols 128:256) and knowledge_hidden
// ---------------------------------------------------------------------------
__global__ __launch_bounds__(512)
void rgcn_kernel(const __bf16* __restrict__ X, int ldx, int x_slice,
                 const __bf16* __restrict__ G, int n_per_b,
                 const __bf16* __restrict__ WrT,
                 const __bf16* __restrict__ WselfT,
                 int mode, __bf16* out1, __bf16* fin2,
                 float* out_f, float* hidden) {
  extern __shared__ __align__(16) char smem[];
  __bf16* sX  = (__bf16*)(smem + RGCN_SX_OFF);    // X compacted to ld=128
  __bf16* sXr = (__bf16*)(smem + RGCN_SXR_OFF);   // Xr^T : (o, j), ld=256
  __bf16* sG  = (__bf16*)(smem + RGCN_SG_OFF);    // gN[b,r] : 256x256, ld=256
  const unsigned sX_lds  = (unsigned)(uintptr_t)sX;
  const unsigned sG_lds  = (unsigned)(uintptr_t)sG;

  const int blk = blockIdx.x;
  const int tid = threadIdx.x;
  const int lane = tid & 31;
  const int wv = tid >> 5;                        // 0..15
  const int n_ = lane & 15, g = lane >> 4;
  const __bf16* Xb = X + (size_t)blk * x_slice;
  const __bf16* Gb = G + (size_t)(blk / n_per_b) * (RR * KK * KK);

  // stage X into LDS (4096 x 16B chunks, 8 per thread), compacting to ld=128
#pragma unroll
  for (int it = 0; it < 8; ++it) {
    const int c = tid + it * 512;
    const int row = c >> 4, o16 = c & 15;
    async_copy_b128(sX_lds + (unsigned)(row * 256 + o16 * 16),
                    (const char*)Xb + (size_t)row * ldx * 2 + o16 * 16);
  }

  v8f acc[8];
  v8f zz = {};
#pragma unroll
  for (int q = 0; q < 8; ++q) acc[q] = zz;

  for (int r = 0; r < RR; ++r) {
    // kick off async prefetch of gN[b,r] (8192 x 16B chunks, 16 per thread)
    const __bf16* Gr = Gb + (size_t)r * KK * KK;
#pragma unroll
    for (int it = 0; it < 16; ++it) {
      const int c = tid + it * 512;
      async_copy_b128(sG_lds + (unsigned)(c * 16), (const char*)Gr + c * 16);
    }
    if (r == 0) {
      // async ops retire in order: cnt<=16 -> the 8 sX chunks are done while
      // the 16 sG chunks stay in flight.
      WAIT_ASYNC_16();
      __syncthreads();
    }
    // stage 1: Xr = X @ Wr[r]   (A from LDS, store transposed into LDS)
    const __bf16* Bt = WrT + (size_t)r * 128 * 128;
#pragma unroll
    for (int q = 0; q < 8; ++q) {
      const int t = wv * 8 + q, tj = t >> 3, to = t & 7;
      v8f a1 = zz;
      for (int k0 = 0; k0 < 128; k0 += 32) {
        v16bf af = load_frag(sX + (size_t)tj * 16 * 128 + k0, 128, lane);
        v16bf bf = load_frag(Bt + (size_t)to * 16 * 128 + k0, 128, lane);
        a1 = wmma_bf16(af, bf, a1);
      }
#pragma unroll
      for (int e = 0; e < 8; ++e)
        sXr[(to * 16 + n_) * 256 + tj * 16 + e + 8 * g] = (__bf16)a1[e];
    }
    WAIT_ASYNC_ALL();        // gN[b,r] resident in LDS (own writes done)
    __syncthreads();         // ... and everyone else's
    // stage 2: acc += gN[b,r] @ Xr   (both operands from LDS)
#pragma unroll
    for (int q = 0; q < 8; ++q) {
      const int t = wv * 8 + q, ti = t >> 3, to = t & 7;
      v8f a2 = acc[q];
      for (int k0 = 0; k0 < 256; k0 += 32) {
        v16bf af = load_frag(sG + (size_t)ti * 16 * 256 + k0, 256, lane);
        v16bf bf = load_frag(sXr + (size_t)to * 16 * 256 + k0, 256, lane);
        a2 = wmma_bf16(af, bf, a2);
      }
      acc[q] = a2;
    }
    __syncthreads();         // protect sXr/sG before next iteration's writes
  }
  // self term: acc += X @ Wself  (A from LDS)
#pragma unroll
  for (int q = 0; q < 8; ++q) {
    const int t = wv * 8 + q, ti = t >> 3, to = t & 7;
    v8f a2 = acc[q];
    for (int k0 = 0; k0 < 128; k0 += 32) {
      v16bf af = load_frag(sX + (size_t)ti * 16 * 128 + k0, 128, lane);
      v16bf bf = load_frag(WselfT + (size_t)to * 16 * 128 + k0, 128, lane);
      a2 = wmma_bf16(af, bf, a2);
    }
    acc[q] = a2;
  }
  // epilogue: relu + scatter
#pragma unroll
  for (int q = 0; q < 8; ++q) {
    const int t = wv * 8 + q, ti = t >> 3, to = t & 7;
#pragma unroll
    for (int e = 0; e < 8; ++e) {
      const int i = ti * 16 + e + 8 * g;
      const int o = to * 16 + n_;
      float v = acc[q][e];
      v = v > 0.f ? v : 0.f;
      if (mode == 1) {
        const int b = blk >> 3, nn = blk & 7;
        out1[((size_t)blk * 256 + i) * 256 + 128 + o] = (__bf16)v;
        fin2[(((size_t)(b * 256 + i)) * 8 + nn) * FIN2_LD + 128 + o] = (__bf16)v;
      } else {
        const int b = blk;
        out_f[((size_t)(b * 256 + i)) * 256 + 128 + o] = v;
        if (i == 0) hidden[b * 128 + o] = v;
      }
    }
  }
}

// ---------------------------------------------------------------------------
// MLP attention:  s[l] = v . tanh(hq[k]+hm[l]); mask; softmax over L=32 lanes;
// out = p @ M.  One wave per (bn,k) pair, lane == l.  Writes bf16 into fin.
// ---------------------------------------------------------------------------
__global__ __launch_bounds__(256)
void attn_kernel(const float* __restrict__ hq, int hq_use_bn,
                 const float* __restrict__ hm, const float* __restrict__ vv,
                 const int* __restrict__ context,
                 const float* __restrict__ M, int ldm, int MO,
                 __bf16* __restrict__ outp, int ldo, int ocol) {
  __shared__ float sP[8 * 32];
  const int lane = threadIdx.x & 31, wv = threadIdx.x >> 5;
  const int pair = blockIdx.x * 8 + wv;
  const int bn = pair >> 8, k = pair & 255;
  const int b = bn >> 3, nn = bn & 7;
  const float* hqp = hq + (size_t)((hq_use_bn ? bn : b) * 256 + k) * 128;
  const float* hmp = hm + (size_t)(bn * 32 + lane) * 128;
  float s = 0.f;
  for (int h = 0; h < 128; ++h) s += vv[h] * tanhf(hqp[h] + hmp[h]);
  if (context[bn * 32 + lane] == 0) s = -1e9f;
  float mx = s;
  for (int off = 16; off; off >>= 1) mx = fmaxf(mx, __shfl_xor(mx, off, 32));
  float e = __expf(s - mx);
  float sum = e;
  for (int off = 16; off; off >>= 1) sum += __shfl_xor(sum, off, 32);
  sP[wv * 32 + lane] = e / sum;        // same-wave LDS, in-order per wave
  __bf16* op = outp + ((size_t)(b * 256 + k) * 8 + nn) * ldo + ocol;
  const float* mp = M + (size_t)bn * 32 * ldm;
  for (int o = lane; o < MO; o += 32) {
    float a = 0.f;
#pragma unroll
    for (int l2 = 0; l2 < 32; ++l2) a += sP[wv * 32 + l2] * mp[l2 * ldm + o];
    op[o] = (__bf16)a;
  }
}

// ---------------------------------------------------------------------------
// Small helper kernels
// ---------------------------------------------------------------------------
__global__ void cvt_kernel(const float* __restrict__ src, __bf16* __restrict__ dst,
                           int R, int C, int ldd, int trans, int total) {
  int i = blockIdx.x * blockDim.x + threadIdx.x;
  if (i >= total) return;
  if (trans == 0) {
    int r = i / ldd, c = i % ldd;
    dst[i] = (__bf16)(c < C ? src[(size_t)r * C + c] : 0.f);
  } else {
    int c = i / ldd, r = i % ldd;
    dst[i] = (__bf16)(r < R ? src[(size_t)r * C + c] : 0.f);
  }
}

// WrT[r][o][d] = sum_nb coef[r,nb] * bases[nb][d][o]
__global__ void wr_kernel(const float* __restrict__ coef,
                          const float* __restrict__ bases,
                          __bf16* __restrict__ WrT) {
  const int r = blockIdx.y;
  const int i = blockIdx.x * blockDim.x + threadIdx.x;   // o*128 + d
  const int o = i >> 7, d = i & 127;
  float s = 0.f;
  for (int nb = 0; nb < NBB; ++nb)
    s += coef[r * NBB + nb] * bases[((size_t)nb * 128 + d) * 128 + o];
  WrT[((size_t)r * 128 + o) * 128 + d] = (__bf16)s;
}

// gN = graph / clip(rowsum, 1.0), written as bf16
__global__ void gnorm_kernel(const float* __restrict__ graph,
                             __bf16* __restrict__ gbf) {
  __shared__ float red[256];
  const int row = blockIdx.x;            // (b,r,i)
  const int t = threadIdx.x;             // j
  float v = graph[(size_t)row * 256 + t];
  red[t] = v;
  __syncthreads();
  for (int s = 128; s > 0; s >>= 1) {
    if (t < s) red[t] += red[t + s];
    __syncthreads();
  }
  float sum = red[0];
  if (sum < 1.f) sum = 1.f;
  gbf[(size_t)row * 256 + t] = (__bf16)(v / sum);
}

__global__ void kbe_kernel(const int* __restrict__ kb_ent,
                           const float* __restrict__ emb,
                           __bf16* __restrict__ kbe) {
  const int row = blockIdx.x;            // b*256+k
  const int c = threadIdx.x;             // 0..127
  kbe[(size_t)row * 128 + c] = (__bf16)emb[(size_t)kb_ent[row] * 128 + c];
}

// fin1 row = (b*256+k)*8 + n : cols 0:128 kb_e, 256 kb_f, 257:288 zero
__global__ void fillfin1_kernel(const __bf16* __restrict__ kbe,
                                const float* __restrict__ kb_f,
                                __bf16* __restrict__ fin1) {
  const int row = blockIdx.x;            // 0..16383
  const int c = threadIdx.x;             // 0..127
  const int bk = row >> 3, nn = row & 7;
  const int b = bk >> 8, k = bk & 255;
  __bf16* dst = fin1 + (size_t)row * FIN1_LD;
  dst[c] = kbe[(size_t)bk * 128 + c];
  if (c < 32)
    dst[256 + c] = (__bf16)((c == 0) ? kb_f[(size_t)(b * NN_ + nn) * KK + k] : 0.f);
}

// fin2: col 512 kb_f, 513:544 zero
__global__ void fillfin2_kernel(const float* __restrict__ kb_f,
                                __bf16* __restrict__ fin2) {
  const int row = blockIdx.x;
  const int c = threadIdx.x;             // 0..31
  const int bk = row >> 3, nn = row & 7;
  const int b = bk >> 8, k = bk & 255;
  fin2[(size_t)row * FIN2_LD + 512 + c] =
      (__bf16)((c == 0) ? kb_f[(size_t)(b * NN_ + nn) * KK + k] : 0.f);
}

__global__ void slen_kernel(const int* __restrict__ context, int* __restrict__ idx) {
  const int b = threadIdx.x;
  if (b < BB) {
    int cnt = 0;
    for (int nn = 0; nn < NN_; ++nn) {
      int nz = 0;
      for (int l = 0; l < LL; ++l) nz |= (context[(b * NN_ + nn) * LL + l] != 0);
      cnt += nz;
    }
    int v = cnt - 1;
    idx[b] = v < 0 ? 0 : v;
  }
}

// GRU elementwise step; mode1 scatters h into kb_o1/fin2, mode2 captures flow2
__global__ void gru_elem_kernel(const float* __restrict__ xg,
                                const float* __restrict__ gh,
                                float* __restrict__ h, __bf16* __restrict__ hbf,
                                int t, int mode,
                                __bf16* kbo1, __bf16* fin2,
                                const int* idx, float* flow2, __bf16* flow2bf) {
  const int row = blockIdx.x;            // b*256 + k
  const int c = threadIdx.x;             // 0..127
  const float* xgp = xg + ((size_t)row * 8 + t) * 384;
  const float* ghp = gh + (size_t)row * 384;
  const float hold = h[(size_t)row * 128 + c];
  const float r = 1.f / (1.f + __expf(-(xgp[c] + ghp[c])));
  const float z = 1.f / (1.f + __expf(-(xgp[128 + c] + ghp[128 + c])));
  const float n2 = tanhf(xgp[256 + c] + r * ghp[256 + c]);
  const float hn = (1.f - z) * n2 + z * hold;
  h[(size_t)row * 128 + c] = hn;
  hbf[(size_t)row * 128 + c] = (__bf16)hn;
  const int b = row >> 8, k = row & 255;
  if (mode == 1) {
    kbo1[(((size_t)(b * 8 + t)) * 256 + k) * 256 + c] = (__bf16)hn;
    fin2[((size_t)row * 8 + t) * FIN2_LD + c] = (__bf16)hn;
  } else {
    if (t == idx[b]) {
      flow2[(size_t)row * 128 + c] = hn;
      flow2bf[(size_t)row * 128 + c] = (__bf16)hn;
    }
  }
}

__global__ void zerof_kernel(float* p, int n) {
  int i = blockIdx.x * 256 + threadIdx.x;
  if (i < n) p[i] = 0.f;
}
__global__ void zerobf_kernel(__bf16* p, int n) {
  int i = blockIdx.x * 256 + threadIdx.x;
  if (i < n) p[i] = (__bf16)0.f;
}

// knowledge_outputs cols 0:128 = flow2
__global__ void outcopy_kernel(const float* __restrict__ flow2,
                               float* __restrict__ outp) {
  const int row = blockIdx.x;            // 0..2047
  const int c = threadIdx.x;             // 0..127
  outp[(size_t)row * 256 + c] = flow2[(size_t)row * 128 + c];
}

// ---------------------------------------------------------------------------
// Host orchestration
// ---------------------------------------------------------------------------
extern "C" void kernel_launch(void* const* d_in, const int* in_sizes, int n_in,
                              void* d_out, int out_size, void* d_ws, size_t ws_size,
                              hipStream_t stream) {
  (void)in_sizes; (void)n_in; (void)out_size; (void)ws_size;
  const int*   context = (const int*)  d_in[0];
  const float* ctx_emb = (const float*)d_in[1];
  const float* ctx_out = (const float*)d_in[2];
  const int*   kb_ent  = (const int*)  d_in[3];
  const float* kb_f    = (const float*)d_in[6];
  const float* graph   = (const float*)d_in[8];
  const float* emb     = (const float*)d_in[9];
  const float* Wq1 = (const float*)d_in[10]; const float* bq1 = (const float*)d_in[11];
  const float* Wm1 = (const float*)d_in[12]; const float* bm1 = (const float*)d_in[13];
  const float* v1  = (const float*)d_in[14];
  const float* Wih1 = (const float*)d_in[15]; const float* Whh1 = (const float*)d_in[16];
  const float* bih1 = (const float*)d_in[17]; const float* bhh1 = (const float*)d_in[18];
  const float* bases1 = (const float*)d_in[19]; const float* coef1 = (const float*)d_in[20];
  const float* Wself1 = (const float*)d_in[21];
  const float* Wq2 = (const float*)d_in[22]; const float* bq2 = (const float*)d_in[23];
  const float* Wm2 = (const float*)d_in[24]; const float* bm2 = (const float*)d_in[25];
  const float* v2  = (const float*)d_in[26];
  const float* Wih2 = (const float*)d_in[27]; const float* Whh2 = (const float*)d_in[28];
  const float* bih2 = (const float*)d_in[29]; const float* bhh2 = (const float*)d_in[30];
  const float* bases2 = (const float*)d_in[31]; const float* coef2 = (const float*)d_in[32];
  const float* Wself2 = (const float*)d_in[33];
  float* out = (float*)d_out;
  float* hidden = out + (size_t)BB * KK * 256;   // 524288

  // allow 256KB dynamic LDS for the fused RGCN kernel (CDNA5 WGP has 320KB)
  hipFuncSetAttribute((const void*)rgcn_kernel,
                      hipFuncAttributeMaxDynamicSharedMemorySize, RGCN_LDS_SIZE);

  char* ws = (char*)d_ws;
  size_t off = 0;
  auto alloc = [&](size_t bytes) -> char* {
    char* p = ws + off;
    off = (off + bytes + 255) & ~(size_t)255;
    return p;
  };
  __bf16* gN_bf   = (__bf16*)alloc((size_t)BB * RR * KK * KK * 2);
  __bf16* kbe_bf  = (__bf16*)alloc((size_t)BK * 128 * 2);
  __bf16* ce_bf   = (__bf16*)alloc((size_t)BN * LL * 128 * 2);
  __bf16* co_bf   = (__bf16*)alloc((size_t)BN * LL * 256 * 2);
  __bf16* Wq1T    = (__bf16*)alloc(128 * 128 * 2);
  __bf16* Wm1T    = (__bf16*)alloc(128 * 128 * 2);
  __bf16* Wq2T    = (__bf16*)alloc(128 * 256 * 2);
  __bf16* Wm2T    = (__bf16*)alloc(128 * 256 * 2);
  __bf16* Wih1P   = (__bf16*)alloc(384 * FIN1_LD * 2);
  __bf16* Whh1P   = (__bf16*)alloc(384 * 128 * 2);
  __bf16* Wih2P   = (__bf16*)alloc(384 * FIN2_LD * 2);
  __bf16* Whh2P   = (__bf16*)alloc(384 * 128 * 2);
  __bf16* Wself1T = (__bf16*)alloc(128 * 128 * 2);
  __bf16* Wself2T = (__bf16*)alloc(128 * 128 * 2);
  __bf16* Wr1T    = (__bf16*)alloc((size_t)RR * 128 * 128 * 2);
  __bf16* Wr2T    = (__bf16*)alloc((size_t)RR * 128 * 128 * 2);
  float*  hq1     = (float*) alloc((size_t)BK * 128 * 4);
  float*  hm1     = (float*) alloc((size_t)BN * LL * 128 * 4);
  float*  hq2     = (float*) alloc((size_t)BN * KK * 128 * 4);
  float*  hm2     = (float*) alloc((size_t)BN * LL * 128 * 4);
  __bf16* fin1    = (__bf16*)alloc((size_t)BK * NN_ * FIN1_LD * 2);
  __bf16* fin2    = (__bf16*)alloc((size_t)BK * NN_ * FIN2_LD * 2);
  float*  xg      = (float*) alloc((size_t)BK * NN_ * 384 * 4);   // shared GRU1/GRU2
  float*  gh      = (float*) alloc((size_t)BK * 384 * 4);
  float*  h       = (float*) alloc((size_t)BK * 128 * 4);
  __bf16* h_bf    = (__bf16*)alloc((size_t)BK * 128 * 2);
  __bf16* kbo1    = (__bf16*)alloc((size_t)BN * KK * 256 * 2);
  float*  flow2   = (float*) alloc((size_t)BK * 128 * 4);
  __bf16* flow2bf = (__bf16*)alloc((size_t)BK * 128 * 2);
  int*    idx     = (int*)   alloc(BB * 4);

  auto cvt = [&](const float* s, __bf16* d, int R, int C, int ldd, int tr) {
    int total = (tr ? C : R) * ldd;
    cvt_kernel<<<(total + 255) / 256, 256, 0, stream>>>(s, d, R, C, ldd, tr, total);
  };
  auto gemm = [&](const __bf16* A, int lda, const __bf16* Bt, int ldb,
                  const float* bias, float* C, int ldc, int M, int Nn, int Kd) {
    gemm_bf16_kernel<<<dim3(Nn / 16, M / 16), 32, 0, stream>>>(A, lda, Bt, ldb,
                                                               bias, C, ldc, Kd);
  };

  // --- preprocessing -------------------------------------------------------
  gnorm_kernel<<<BB * RR * KK, 256, 0, stream>>>(graph, gN_bf);
  kbe_kernel<<<BK, 128, 0, stream>>>(kb_ent, emb, kbe_bf);
  cvt(ctx_emb, ce_bf, BN * LL, 128, 128, 0);
  cvt(ctx_out, co_bf, BN * LL, 256, 256, 0);
  cvt(Wq1, Wq1T, 128, 128, 128, 1);
  cvt(Wm1, Wm1T, 128, 128, 128, 1);
  cvt(Wq2, Wq2T, 256, 128, 256, 1);
  cvt(Wm2, Wm2T, 256, 128, 256, 1);
  cvt(Wih1, Wih1P, 384, 257, FIN1_LD, 0);
  cvt(Whh1, Whh1P, 384, 128, 128, 0);
  cvt(Wih2, Wih2P, 384, 513, FIN2_LD, 0);
  cvt(Whh2, Whh2P, 384, 128, 128, 0);
  cvt(Wself1, Wself1T, 128, 128, 128, 1);
  cvt(Wself2, Wself2T, 128, 128, 128, 1);
  wr_kernel<<<dim3(64, RR), 256, 0, stream>>>(coef1, bases1, Wr1T);
  wr_kernel<<<dim3(64, RR), 256, 0, stream>>>(coef2, bases2, Wr2T);
  slen_kernel<<<1, 32, 0, stream>>>(context, idx);

  // --- attention 1 ---------------------------------------------------------
  gemm(kbe_bf, 128, Wq1T, 128, bq1, hq1, 128, BK, 128, 128);         // hq1 (b,k)
  gemm(ce_bf, 128, Wm1T, 128, bm1, hm1, 128, BN * LL, 128, 128);     // hm1
  fillfin1_kernel<<<BK * NN_, 128, 0, stream>>>(kbe_bf, kb_f, fin1);
  attn_kernel<<<(BN * KK) / 8, 256, 0, stream>>>(hq1, 0, hm1, v1, context,
                                                 ctx_emb, 128, 128,
                                                 fin1, FIN1_LD, 128);

  // --- GRU 1 ---------------------------------------------------------------
  gemm(fin1, FIN1_LD, Wih1P, FIN1_LD, bih1, xg, 384, BK * NN_, 384, FIN1_LD);
  zerof_kernel<<<(BK * 128 + 255) / 256, 256, 0, stream>>>(h, BK * 128);
  zerobf_kernel<<<(BK * 128 + 255) / 256, 256, 0, stream>>>(h_bf, BK * 128);
  for (int t = 0; t < NN_; ++t) {
    gemm(h_bf, 128, Whh1P, 128, bhh1, gh, 384, BK, 384, 128);
    gru_elem_kernel<<<BK, 128, 0, stream>>>(xg, gh, h, h_bf, t, 1, kbo1, fin2,
                                            nullptr, nullptr, nullptr);
  }

  // --- RGCN 1 (fused, WMMA + async LDS staging) ---------------------------
  rgcn_kernel<<<BN, 512, RGCN_LDS_SIZE, stream>>>(kbo1, 256, 256 * 256, gN_bf,
                                                  NN_, Wr1T, Wself1T, 1, kbo1,
                                                  fin2, nullptr, nullptr);

  // --- attention 2 ---------------------------------------------------------
  gemm(kbo1, 256, Wq2T, 256, bq2, hq2, 128, BN * KK, 128, 256);
  gemm(co_bf, 256, Wm2T, 256, bm2, hm2, 128, BN * LL, 128, 256);
  fillfin2_kernel<<<BK * NN_, 32, 0, stream>>>(kb_f, fin2);
  attn_kernel<<<(BN * KK) / 8, 256, 0, stream>>>(hq2, 1, hm2, v2, context,
                                                 ctx_out, 256, 256,
                                                 fin2, FIN2_LD, 256);

  // --- GRU 2 ---------------------------------------------------------------
  gemm(fin2, FIN2_LD, Wih2P, FIN2_LD, bih2, xg, 384, BK * NN_, 384, FIN2_LD);
  zerof_kernel<<<(BK * 128 + 255) / 256, 256, 0, stream>>>(h, BK * 128);
  zerobf_kernel<<<(BK * 128 + 255) / 256, 256, 0, stream>>>(h_bf, BK * 128);
  for (int t = 0; t < NN_; ++t) {
    gemm(h_bf, 128, Whh2P, 128, bhh2, gh, 384, BK, 384, 128);
    gru_elem_kernel<<<BK, 128, 0, stream>>>(xg, gh, h, h_bf, t, 2, nullptr,
                                            nullptr, idx, flow2, flow2bf);
  }

  // --- RGCN 2 + outputs ----------------------------------------------------
  rgcn_kernel<<<BB, 512, RGCN_LDS_SIZE, stream>>>(flow2bf, 128, 256 * 128, gN_bf,
                                                  1, Wr2T, Wself2T, 2, nullptr,
                                                  nullptr, out, hidden);
  outcopy_kernel<<<BK, 128, 0, stream>>>(flow2, out);
}
